// KANLayer_53463752900835
// MI455X (gfx1250) — matrix operand order repaired and checked by reading, not compiled, not run on hardware
//
#include <hip/hip_runtime.h>

#define IN_DIM  256
#define OUT_DIM 256
#define NUM     16
#define KORD    3
#define G_PTS   (NUM + 2 * KORD + 1)      // 23 grid points per input dim
#define NUMK    (NUM + KORD)              // 19 spline bases per input dim
#define BATCH   2048
#define KTOT    (IN_DIM + IN_DIM * NUMK)  // 256 + 4864 = 5120 fused reduction dim
#define KSTEPS  (KTOT / 32)               // 160 WMMA k-steps

typedef __attribute__((ext_vector_type(16))) __bf16 v16bf;
typedef __attribute__((ext_vector_type(8)))  __bf16 v8bf;
typedef __attribute__((ext_vector_type(8)))  float  v8f;

union Frag {
    v16bf v;
    struct { v8bf lo; v8bf hi; } h;
};

struct Stage {
    Frag a0, a1, b0, b1;
};

// ---------------------------------------------------------------------------
// Fold scale_base/scale_sp/mask/coef into a transposed bf16 weight matrix
// Wt[o, kk], kk in [0,5120):  kk<256 -> base weights, else spline weights.
// One thread per (i, o) pair.
// ---------------------------------------------------------------------------
__global__ __launch_bounds__(256) void kan_prep_w(
    const float* __restrict__ coef,
    const float* __restrict__ scale_base,
    const float* __restrict__ scale_sp,
    const float* __restrict__ mask,
    __bf16* __restrict__ Wt)
{
    const int t = blockIdx.x * blockDim.x + threadIdx.x;  // 0..65535
    const int i = t >> 8;
    const int o = t & 255;

    const size_t io = (size_t)i * OUT_DIM + o;
    const float  m  = mask[io];

    __bf16* row = Wt + (size_t)o * KTOT;
    row[i] = (__bf16)(scale_base[io] * m);

    const float  ss = scale_sp[io] * m;
    const float* c  = coef + io * NUMK;
#pragma unroll
    for (int kk = 0; kk < NUMK; ++kk)
        row[IN_DIM + i * NUMK + kk] = (__bf16)(c[kk] * ss);
}

// ---------------------------------------------------------------------------
// Build fused activation matrix Act[b, kk] in bf16:
//   Act[b, i]                 = silu(x[b,i])
//   Act[b, 256 + i*19 + kk]   = B-spline basis kk (degree 3) of x[b,i]
// One thread per (b, i) pair. Cox–de-Boor recursion matches the reference's
// indexing exactly (left/right slicing).
// ---------------------------------------------------------------------------
__global__ __launch_bounds__(256) void kan_prep_act(
    const float* __restrict__ x,
    const float* __restrict__ grid,
    __bf16* __restrict__ Act)
{
    const int t = blockIdx.x * blockDim.x + threadIdx.x;  // 0..BATCH*IN_DIM-1
    const int b = t >> 8;
    const int i = t & 255;

    const float xv = x[(size_t)b * IN_DIM + i];

    __bf16* row = Act + (size_t)b * KTOT;

    // silu(x) = x * sigmoid(x)
    row[i] = (__bf16)(xv / (1.0f + __expf(-xv)));

    float tg[G_PTS];
    const float* gi = grid + (size_t)i * G_PTS;
#pragma unroll
    for (int g = 0; g < G_PTS; ++g) tg[g] = gi[g];

    float val[G_PTS - 1];  // 22 degree-0 indicators
#pragma unroll
    for (int g = 0; g < G_PTS - 1; ++g)
        val[g] = (xv >= tg[g] && xv < tg[g + 1]) ? 1.0f : 0.0f;

#pragma unroll
    for (int j = 1; j <= KORD; ++j) {
#pragma unroll
        for (int g = 0; g < G_PTS - 1 - j; ++g) {
            const float left  = (xv - tg[g]) / (tg[g + j] - tg[g]);
            const float right = (tg[g + j + 1] - xv) / (tg[g + j + 1] - tg[g + 1]);
            val[g] = left * val[g] + right * val[g + 1];
        }
    }

#pragma unroll
    for (int kk = 0; kk < NUMK; ++kk)
        row[IN_DIM + i * NUMK + kk] = (__bf16)val[kk];
}

// ---------------------------------------------------------------------------
// out(2048x256) = Act(2048x5120) x Wt(256x5120)^T using v_wmma_f32_16x16x32_bf16.
// 64x64 block tile, 4 waves in a 2x2 arrangement, each wave owns a 32x32 tile
// (2x2 WMMA accumulators). K loop: 160 steps of 32, explicitly 2-stage
// software-pipelined so the 8 global_load_b128 of step n+1 are in flight while
// the 4 WMMAs of step n execute (waits become loadcnt<=8 instead of 0).
//
// Fragment layouts per CDNA5 ISA 7.12.2 (wave32):
//   A 16x32 bf16: lane&15 = M row; lanes<16 hold K {0..7,16..23},
//                 lanes>=16 hold K {8..15,24..31}  -> two 16B loads (+0,+32B).
//   B 32x16 bf16: lane&15 = N col; lanes<16 hold K {0..7,16..23},
//                 lanes>=16 hold K {8..15,24..31}  -> two 16B loads (+0,+32B).
//   C/D 16x16 f32: VGPR r holds M=r (lanes<16) / M=8+r (lanes>=16), N=lane&15.
// ---------------------------------------------------------------------------
__global__ __launch_bounds__(128) void kan_gemm(
    const __bf16* __restrict__ Act,
    const __bf16* __restrict__ Wt,
    float* __restrict__ out)
{
    const int lane  = threadIdx.x & 31;
    const int wave  = threadIdx.x >> 5;   // 0..3
    const int waveM = wave >> 1;
    const int waveN = wave & 1;
    const int lr    = lane & 15;
    const int sel   = lane >> 4;          // which half-wave

    const int rowBase = blockIdx.x * 64 + waveM * 32;
    const int colBase = blockIdx.y * 64 + waveN * 32;

    // Per-lane base pointers for the two M-tiles / two N-tiles of this wave.
    const __bf16* a0p = Act + (size_t)(rowBase + lr)      * KTOT + sel * 8;
    const __bf16* a1p = Act + (size_t)(rowBase + 16 + lr) * KTOT + sel * 8;
    const __bf16* b0p = Wt  + (size_t)(colBase + lr)      * KTOT + sel * 8;
    const __bf16* b1p = Wt  + (size_t)(colBase + 16 + lr) * KTOT + sel * 8;

    v8f acc00 = {}, acc01 = {}, acc10 = {}, acc11 = {};

    Stage st[2];

    auto loadStage = [&](Stage& s, int k0) {
        s.a0.h.lo = *(const v8bf*)(a0p + k0);
        s.a0.h.hi = *(const v8bf*)(a0p + k0 + 16);
        s.a1.h.lo = *(const v8bf*)(a1p + k0);
        s.a1.h.hi = *(const v8bf*)(a1p + k0 + 16);
        s.b0.h.lo = *(const v8bf*)(b0p + k0);
        s.b0.h.hi = *(const v8bf*)(b0p + k0 + 16);
        s.b1.h.lo = *(const v8bf*)(b1p + k0);
        s.b1.h.hi = *(const v8bf*)(b1p + k0 + 16);
    };

    loadStage(st[0], 0);

#pragma unroll 2
    for (int ks = 0; ks < KSTEPS; ++ks) {
        const int k0 = ks * 32;

        // Issue next stage's loads before consuming the current stage.
        if (ks + 1 < KSTEPS)
            loadStage(st[(ks + 1) & 1], k0 + 32);

        // Keep L2 primed two stages ahead (global_prefetch_b8).
        __builtin_prefetch(a0p + k0 + 64, 0, 1);
        __builtin_prefetch(a1p + k0 + 64, 0, 1);
        __builtin_prefetch(b0p + k0 + 64, 0, 1);
        __builtin_prefetch(b1p + k0 + 64, 0, 1);

        const Stage& c = st[ks & 1];
        acc00 = __builtin_amdgcn_wmma_f32_16x16x32_bf16(false, c.a0.v, false, c.b0.v,
                                                        (short)0, acc00, false, false);
        acc01 = __builtin_amdgcn_wmma_f32_16x16x32_bf16(false, c.a0.v, false, c.b1.v,
                                                        (short)0, acc01, false, false);
        acc10 = __builtin_amdgcn_wmma_f32_16x16x32_bf16(false, c.a1.v, false, c.b0.v,
                                                        (short)0, acc10, false, false);
        acc11 = __builtin_amdgcn_wmma_f32_16x16x32_bf16(false, c.a1.v, false, c.b1.v,
                                                        (short)0, acc11, false, false);
    }

    // Scatter C/D fragments to f32 output (row stride = OUT_DIM).
#pragma unroll
    for (int r = 0; r < 8; ++r) {
        const int row0 = rowBase + sel * 8 + r;
        out[(size_t)row0 * OUT_DIM        + colBase + lr]      = acc00[r];
        out[(size_t)row0 * OUT_DIM        + colBase + 16 + lr] = acc01[r];
        out[(size_t)(row0 + 16) * OUT_DIM + colBase + lr]      = acc10[r];
        out[(size_t)(row0 + 16) * OUT_DIM + colBase + 16 + lr] = acc11[r];
    }
}

// ---------------------------------------------------------------------------
extern "C" void kernel_launch(void* const* d_in, const int* in_sizes, int n_in,
                              void* d_out, int out_size, void* d_ws, size_t ws_size,
                              hipStream_t stream)
{
    const float* x          = (const float*)d_in[0];
    const float* grid       = (const float*)d_in[1];
    const float* coef       = (const float*)d_in[2];
    const float* scale_base = (const float*)d_in[3];
    const float* scale_sp   = (const float*)d_in[4];
    const float* mask       = (const float*)d_in[5];
    // d_in[6] is the python scalar k (== 3), baked in as KORD.

    float* out = (float*)d_out;

    // Workspace: Act (2048x5120 bf16, ~21 MB) then Wt (256x5120 bf16, ~2.6 MB).
    const size_t actBytes = (size_t)BATCH * KTOT * 2;
    __bf16* Act = (__bf16*)d_ws;
    __bf16* Wt  = (__bf16*)((char*)d_ws + actBytes);

    kan_prep_w  <<<(IN_DIM * OUT_DIM) / 256, 256, 0, stream>>>(coef, scale_base,
                                                               scale_sp, mask, Wt);
    kan_prep_act<<<(BATCH * IN_DIM) / 256, 256, 0, stream>>>(x, grid, Act);

    dim3 g(BATCH / 64, OUT_DIM / 64);   // 32 x 4 = 128 workgroups
    kan_gemm<<<g, 128, 0, stream>>>(Act, Wt, out);
}
